// FloydBlock2_49143015801464
// MI455X (gfx1250) — matrix operand order repaired and checked by reading, not compiled, not run on hardware
//
#include <hip/hip_runtime.h>
#include <math.h>

// ---------------------------------------------------------------------------
// Pivotal attention for MI455X (gfx1250, wave32, WMMA + async-to-LDS).
// B=4, N=128, C=512, H=8, D=64.  SCALE = 64^-0.5 * 50 = 6.25
// ---------------------------------------------------------------------------

typedef _Float16 h16;
typedef __attribute__((ext_vector_type(4)))  _Float16 v4h;
typedef __attribute__((ext_vector_type(8)))  _Float16 v8h;
typedef __attribute__((ext_vector_type(16))) _Float16 v16h;
typedef __attribute__((ext_vector_type(4)))  float    v4f;
typedef __attribute__((ext_vector_type(8)))  float    v8f;

#define B_ 4
#define N_ 128
#define C_ 512
#define H_ 8
#define D_ 64
#define FIVE_C 2560

static constexpr float  SCALE = 6.25f;
static constexpr long   NX    = (long)B_ * N_ * N_ * C_;      // 33,554,432 floats in x
static constexpr long   NW    = (long)C_ * FIVE_C;            // 1,310,720 weights
static constexpr int    NSLICE = B_ * H_ * N_;                // 4096 GEMM slices

// ---------------------------------------------------------------------------
// Async global->LDS copy (GLOBAL_LOAD_ASYNC_TO_LDS_B128, GVS mode).
// Per CDNA5 ISA: each lane copies global SADDR+VADDR+off to LDS at its VDST offset,
// 16 bytes per lane, tracked by ASYNCcnt.  LDS operand is the 32-bit LDS byte
// offset (low 32 bits of a generic pointer to __shared__).
// ---------------------------------------------------------------------------
__device__ __forceinline__ void async_copy_b128(const h16* ldsDst,
                                                const h16* gBase,
                                                unsigned byteOff) {
  unsigned ldsOff = (unsigned)(size_t)(const void*)ldsDst;
  asm volatile("global_load_async_to_lds_b128 %0, %1, %2"
               :
               : "v"(ldsOff), "v"(byteOff), "s"(gBase)
               : "memory");
}

__device__ __forceinline__ void wait_async() {
  asm volatile("s_wait_asynccnt 0x0" ::: "memory");
}

// ---------------------------------------------------------------------------
// WMMA fragment loaders (wave32, 16x16x32 f16).
// A 16x32 (M x K): lane L holds row M=L%16; lanes<16 hold K {0..7,16..23},
//                  lanes>=16 hold K {8..15,24..31}; v16h elem e -> VGPR e/2.
// B 32x16 (K x N): lane L holds col N=L%16; lanes<16 hold K {0..7,16..23},
//                  lanes>=16 hold K {8..15,24..31} (same order as A).
// C/D 16x16 f32:   lane L col N=L%16; VGPR r -> row M = r + (L<16 ? 0 : 8).
// ---------------------------------------------------------------------------
__device__ __forceinline__ v16h load_frag_a(const h16* lds, int ldm, int row,
                                            int kBase, int lane) {
  const h16* p = lds + (size_t)row * ldm + kBase + ((lane & 16) ? 8 : 0);
  v8h lo = *(const v8h*)p;          // K offsets +0..+7   (rel to lane half)
  v8h hi = *(const v8h*)(p + 16);   // K offsets +16..+23
  return __builtin_shufflevector(lo, hi, 0, 1, 2, 3, 4, 5, 6, 7,
                                         8, 9, 10, 11, 12, 13, 14, 15);
}

__device__ __forceinline__ v16h load_frag_b(const h16* lds, int ldm, int col,
                                            int kBase, int lane) {
  const h16* p = lds + (size_t)col * ldm + kBase + ((lane & 16) ? 8 : 0);
  v8h lo = *(const v8h*)p;          // K +0..+7
  v8h hi = *(const v8h*)(p + 16);   // K +16..+23
  return __builtin_shufflevector(lo, hi, 0, 1, 2, 3, 4, 5, 6, 7,
                                         8, 9, 10, 11, 12, 13, 14, 15);
}

#define WMMA_F32_F16(a, b, c) \
  __builtin_amdgcn_wmma_f32_16x16x32_f16(false, (a), false, (b), (short)0, (c), false, false)

// ---------------------------------------------------------------------------
// Stage 1: deterministic global mean / std reduction (two stages, no atomics)
// ---------------------------------------------------------------------------
__global__ __launch_bounds__(256) void reduce_partials_kernel(
    const float* __restrict__ x, float* __restrict__ partials) {
  __shared__ float red[512];
  const int t = threadIdx.x;
  float s = 0.f, q = 0.f;
  const long n4 = NX / 4;
  for (long idx = (long)blockIdx.x * 256 + t; idx < n4; idx += (long)gridDim.x * 256) {
    v4f v = *(const v4f*)(x + idx * 4);
#pragma unroll
    for (int e = 0; e < 4; e++) { s += v[e]; q += v[e] * v[e]; }
  }
  red[t] = s; red[256 + t] = q;
  __syncthreads();
  for (int st = 128; st > 0; st >>= 1) {
    if (t < st) { red[t] += red[t + st]; red[256 + t] += red[256 + t + st]; }
    __syncthreads();
  }
  if (t == 0) { partials[2 * blockIdx.x] = red[0]; partials[2 * blockIdx.x + 1] = red[256]; }
}

__global__ __launch_bounds__(256) void finalize_stats_kernel(
    const float* __restrict__ partials, float* __restrict__ sc) {
  __shared__ float red[512];
  const int t = threadIdx.x;
  float s = 0.f, q = 0.f;
#pragma unroll
  for (int r = 0; r < 4; r++) { s += partials[2 * (t + 256 * r)]; q += partials[2 * (t + 256 * r) + 1]; }
  red[t] = s; red[256 + t] = q;
  __syncthreads();
  for (int st = 128; st > 0; st >>= 1) {
    if (t < st) { red[t] += red[t + st]; red[256 + t] += red[256 + t + st]; }
    __syncthreads();
  }
  if (t == 0) {
    double sum = red[0], sq = red[256];
    double n = (double)NX;
    double mean = sum / n;
    double var = (sq - n * mean * mean) / (n - 1.0);   // unbiased (ddof=1)
    sc[2] = (float)mean;
    sc[3] = (float)(1.0 / sqrt(var));
  }
}

// ---------------------------------------------------------------------------
// Stage 2: normalize x -> f16, convert weights -> f16
// ---------------------------------------------------------------------------
__global__ __launch_bounds__(256) void norm_convert_kernel(
    const float* __restrict__ x, const float* __restrict__ sc,
    h16* __restrict__ xn) {
  const float mean = sc[2], inv = sc[3];
  const long idx = ((long)blockIdx.x * 256 + threadIdx.x) * 4;
  v4f v = *(const v4f*)(x + idx);
  v4h o;
#pragma unroll
  for (int e = 0; e < 4; e++) o[e] = (h16)((v[e] - mean) * inv);
  *(v4h*)(xn + idx) = o;
}

__global__ __launch_bounds__(256) void convert_w_kernel(
    const float* __restrict__ w, h16* __restrict__ wh) {
  const long idx = ((long)blockIdx.x * 256 + threadIdx.x) * 4;
  v4f v = *(const v4f*)(w + idx);
  v4h o;
#pragma unroll
  for (int e = 0; e < 4; e++) o[e] = (h16)v[e];
  *(v4h*)(wh + idx) = o;
}

// ---------------------------------------------------------------------------
// Stage 3: QKV GEMM  [65536 x 512] @ [512 x 2560] -> f16 qkv
// WG tile 128x128, 8 waves, each wave a 16x128 strip; K staged 32 at a time.
// A tile staged via async-to-LDS; B tile transposed via VGPR scatter.
// ---------------------------------------------------------------------------
__global__ __launch_bounds__(256) void qkv_gemm_kernel(
    const h16* __restrict__ xn, const h16* __restrict__ wh,
    h16* __restrict__ qkv) {
  constexpr int LDA = 40;  // 32 + 8 pad (16B-aligned rows)
  constexpr int LDB = 40;
  __shared__ __attribute__((aligned(16))) h16 sA[128 * LDA];
  __shared__ __attribute__((aligned(16))) h16 sB[128 * LDB];  // transposed [n][k]
  const int t = threadIdx.x, lane = t & 31, wave = t >> 5;
  const int mWG = blockIdx.x * 128;
  const int nWG = blockIdx.y * 128;

  v8f zero = {0.f, 0.f, 0.f, 0.f, 0.f, 0.f, 0.f, 0.f};
  v8f acc[8];
#pragma unroll
  for (int i = 0; i < 8; i++) acc[i] = zero;

  const int ar = t >> 1, ac = (t & 1) * 16;           // A copy mapping
  const int bk = t >> 3, bc = (t & 7) * 16;           // B copy mapping
  const h16* aBase = xn + (size_t)mWG * C_;           // uniform SGPR base

  for (int k0 = 0; k0 < C_; k0 += 32) {
    {  // A tile 128x32 : two async b128 per thread, no VGPR round-trip
      const unsigned off = (unsigned)(((size_t)ar * C_ + k0 + ac) * 2);
      async_copy_b128(&sA[ar * LDA + ac],     aBase, off);
      async_copy_b128(&sA[ar * LDA + ac + 8], aBase, off + 16);
      if (k0 + 32 < C_)  // prefetch next A tile into cache
        __builtin_prefetch(aBase + (size_t)ar * C_ + k0 + 32 + ac, 0, 0);
    }
    {  // B tile 32x128, stored transposed for contiguous-K fragment reads
      const h16* g = wh + (size_t)(k0 + bk) * FIVE_C + nWG + bc;
      v8h b0 = *(const v8h*)g;
      v8h b1 = *(const v8h*)(g + 8);
#pragma unroll
      for (int e = 0; e < 8; e++) {
        sB[(bc + e) * LDB + bk]     = b0[e];
        sB[(bc + 8 + e) * LDB + bk] = b1[e];
      }
    }
    wait_async();
    __syncthreads();

    const int mRow = wave * 16 + (lane & 15);
    v16h af = load_frag_a(sA, LDA, mRow, 0, lane);
    v16h bf[8];
#pragma unroll
    for (int nt = 0; nt < 8; nt++)
      bf[nt] = load_frag_b(sB, LDB, nt * 16 + (lane & 15), 0, lane);
#pragma unroll
    for (int nt = 0; nt < 8; nt++)
      acc[nt] = WMMA_F32_F16(af, bf[nt], acc[nt]);
    __syncthreads();
  }

  const int col = lane & 15;
  const int rOff = (lane & 16) ? 8 : 0;
#pragma unroll
  for (int nt = 0; nt < 8; nt++)
#pragma unroll
    for (int r = 0; r < 8; r++) {
      const size_t row = (size_t)(mWG + wave * 16 + r + rOff);
      qkv[row * FIVE_C + nWG + nt * 16 + col] = (h16)acc[nt][r];
    }
}

// ---------------------------------------------------------------------------
// Stage 4: scores.  One WG per slice s = (b*8+h)*128 + x.
// mode 1 (x=i): S1[j,k] = SCALE * Q_i[j,:] . K1_i[k,:]   (row stride 2560)
// mode 2 (x=j): S2[i,k] = SCALE * Q_j[i,:] . K2_j[k,:]   (row stride 128*2560)
// Q/K tiles staged via async-to-LDS.
// ---------------------------------------------------------------------------
__global__ __launch_bounds__(256) void scores_kernel(
    const h16* __restrict__ qkv, h16* __restrict__ sOut, int mode) {
  constexpr int LD = 72;  // 64 + 8 pad
  __shared__ __attribute__((aligned(16))) h16 sQ[128 * LD];
  __shared__ __attribute__((aligned(16))) h16 sK[128 * LD];
  const int t = threadIdx.x, lane = t & 31, wave = t >> 5;
  const int s = blockIdx.x;
  const int x = s & 127, h = (s >> 7) & 7, b = s >> 10;

  size_t aOff, bOff, rStride;
  if (mode == 1) {
    aOff = ((size_t)(b * 16384 + x * 128)) * FIVE_C + h * 64;   // q rows (j)
    bOff = aOff + C_;                                           // k1 rows (k)
    rStride = FIVE_C;
  } else {
    aOff = ((size_t)(b * 16384 + x)) * FIVE_C + h * 64;         // q rows (i)
    bOff = aOff + 2 * C_;                                       // k2 rows (k)
    rStride = (size_t)N_ * FIVE_C;
  }

  {  // stage Q and K tiles (128 x 64 each): 4 async b128 per thread each
    const int r = t >> 1, c = (t & 1) * 32;
    const h16* gq = qkv + aOff;   // uniform bases
    const h16* gk = qkv + bOff;
    const unsigned off0 = (unsigned)(((size_t)r * rStride + c) * 2);
#pragma unroll
    for (int q = 0; q < 4; q++) {
      async_copy_b128(&sQ[r * LD + c + q * 8], gq, off0 + q * 16);
      async_copy_b128(&sK[r * LD + c + q * 8], gk, off0 + q * 16);
    }
  }
  wait_async();
  __syncthreads();

  v8f zero = {0.f, 0.f, 0.f, 0.f, 0.f, 0.f, 0.f, 0.f};
  v8f acc[8];
#pragma unroll
  for (int i = 0; i < 8; i++) acc[i] = zero;

  const int mRow = wave * 16 + (lane & 15);
#pragma unroll
  for (int ks = 0; ks < 2; ks++) {
    v16h af = load_frag_a(sQ, LD, mRow, ks * 32, lane);
    v16h bf[8];
#pragma unroll
    for (int nt = 0; nt < 8; nt++)
      bf[nt] = load_frag_b(sK, LD, nt * 16 + (lane & 15), ks * 32, lane);
#pragma unroll
    for (int nt = 0; nt < 8; nt++)
      acc[nt] = WMMA_F32_F16(af, bf[nt], acc[nt]);
  }

  h16* out = sOut + (size_t)s * 16384;
  const int col = lane & 15;
  const int rOff = (lane & 16) ? 8 : 0;
#pragma unroll
  for (int nt = 0; nt < 8; nt++)
#pragma unroll
    for (int r = 0; r < 8; r++)
      out[(wave * 16 + r + rOff) * 128 + nt * 16 + col] = (h16)(acc[nt][r] * SCALE);
}

// ---------------------------------------------------------------------------
// Stage 5: softmax over k, in place into BOTH score layouts (1:1 row mapping).
// One wave per (b,h,i,j) row; 4 elements per lane, shuffle reductions.
// ---------------------------------------------------------------------------
__global__ __launch_bounds__(256) void softmax_kernel(
    h16* __restrict__ s1, h16* __restrict__ s2) {
  const int t = threadIdx.x, lane = t & 31, w = t >> 5;
  const long row = (long)blockIdx.x * 8 + w;
  const int j = (int)(row & 127);
  const int i = (int)((row >> 7) & 127);
  const int h = (int)((row >> 14) & 7);
  const int b = (int)(row >> 17);
  h16* r1 = s1 + ((size_t)((b * 8 + h) * 128 + i)) * 16384 + j * 128;
  h16* r2 = s2 + ((size_t)((b * 8 + h) * 128 + j)) * 16384 + i * 128;

  v4h a = *(const v4h*)(r1 + lane * 4);
  v4h c = *(const v4h*)(r2 + lane * 4);
  float v[4];
#pragma unroll
  for (int e = 0; e < 4; e++) v[e] = (float)a[e] + (float)c[e];

  float m = fmaxf(fmaxf(v[0], v[1]), fmaxf(v[2], v[3]));
#pragma unroll
  for (int off = 16; off > 0; off >>= 1) m = fmaxf(m, __shfl_xor(m, off, 32));

  float e0 = __expf(v[0] - m), e1 = __expf(v[1] - m);
  float e2 = __expf(v[2] - m), e3 = __expf(v[3] - m);
  float sum = e0 + e1 + e2 + e3;
#pragma unroll
  for (int off = 16; off > 0; off >>= 1) sum += __shfl_xor(sum, off, 32);
  const float inv = 1.f / sum;

  v4h o;
  o[0] = (h16)(e0 * inv); o[1] = (h16)(e1 * inv);
  o[2] = (h16)(e2 * inv); o[3] = (h16)(e3 * inv);
  *(v4h*)(r1 + lane * 4) = o;
  *(v4h*)(r2 + lane * 4) = o;
}

// ---------------------------------------------------------------------------
// Stage 6: out = attn @ V   (128x128 @ 128x64 per slice)
// mode 1 (x=i): O[j,d] = A[j,k] V1_i[k,d]   -> store (first writer)
// mode 2 (x=j): O[i,d] = A'[i,k] V2_j[k,d]  -> accumulate
// attn tile staged via async-to-LDS; V tile transposed via VGPR scatter.
// ---------------------------------------------------------------------------
__global__ __launch_bounds__(256) void out_kernel(
    const h16* __restrict__ qkv, const h16* __restrict__ attn,
    float* __restrict__ out, int mode) {
  constexpr int LDA2 = 136;  // 128 + 8 pad
  constexpr int LDV  = 136;
  __shared__ __attribute__((aligned(16))) h16 sAt[128 * LDA2];
  __shared__ __attribute__((aligned(16))) h16 sV[64 * LDV];   // transposed [d][k]
  const int t = threadIdx.x, lane = t & 31, wave = t >> 5;
  const int s = blockIdx.x;
  const int x = s & 127, h = (s >> 7) & 7, b = s >> 10;

  size_t vOff, vStride, outBase, outStride;
  if (mode == 1) {
    vOff = ((size_t)(b * 16384 + x * 128)) * FIVE_C + 3 * C_ + h * 64;  // v1 rows k
    vStride = FIVE_C;
    outBase = ((size_t)(b * 16384 + x * 128)) * C_ + h * 64;            // rows j
    outStride = C_;
  } else {
    vOff = ((size_t)(b * 16384 + x)) * FIVE_C + 4 * C_ + h * 64;        // v2 rows k
    vStride = (size_t)N_ * FIVE_C;
    outBase = ((size_t)b * 16384) * C_ + (size_t)x * C_ + h * 64;       // rows i
    outStride = (size_t)N_ * C_;
  }

  const h16* aG = attn + (size_t)s * 16384;
  {  // stage attn 128x128 : 8 async b128 per thread
    const int r = t >> 1, c = (t & 1) * 64;
    const unsigned off0 = (unsigned)(((size_t)r * 128 + c) * 2);
#pragma unroll
    for (int q = 0; q < 8; q++)
      async_copy_b128(&sAt[r * LDA2 + c + q * 8], aG, off0 + q * 16);
  }
  {  // stage V 128x64, transposed -> sV[d][k]
    const int k = t >> 1, c = (t & 1) * 32;
    const h16* g = qkv + vOff + (size_t)k * vStride + c;
#pragma unroll
    for (int q = 0; q < 4; q++) {
      v8h vv = *(const v8h*)(g + q * 8);
#pragma unroll
      for (int e = 0; e < 8; e++) sV[(c + q * 8 + e) * LDV + k] = vv[e];
    }
  }
  wait_async();
  __syncthreads();

  v8f zero = {0.f, 0.f, 0.f, 0.f, 0.f, 0.f, 0.f, 0.f};
  v8f acc[4];
#pragma unroll
  for (int i = 0; i < 4; i++) acc[i] = zero;

  const int mRow = wave * 16 + (lane & 15);
#pragma unroll
  for (int ks = 0; ks < 4; ks++) {
    v16h af = load_frag_a(sAt, LDA2, mRow, ks * 32, lane);
    v16h bf[4];
#pragma unroll
    for (int nt = 0; nt < 4; nt++)
      bf[nt] = load_frag_b(sV, LDV, nt * 16 + (lane & 15), ks * 32, lane);
#pragma unroll
    for (int nt = 0; nt < 4; nt++)
      acc[nt] = WMMA_F32_F16(af, bf[nt], acc[nt]);
  }

  const int col = lane & 15;
  const int rOff = (lane & 16) ? 8 : 0;
#pragma unroll
  for (int nt = 0; nt < 4; nt++)
#pragma unroll
    for (int r = 0; r < 8; r++) {
      const size_t o = outBase + (size_t)(wave * 16 + r + rOff) * outStride + nt * 16 + col;
      if (mode == 1) out[o] = acc[nt][r];
      else           out[o] += acc[nt][r];
    }
}

// ---------------------------------------------------------------------------
// Host launcher
// ---------------------------------------------------------------------------
extern "C" void kernel_launch(void* const* d_in, const int* in_sizes, int n_in,
                              void* d_out, int out_size, void* d_ws, size_t ws_size,
                              hipStream_t stream) {
  (void)in_sizes; (void)n_in; (void)out_size; (void)ws_size;
  const float* x = (const float*)d_in[0];
  const float* w = (const float*)d_in[1];
  float* out = (float*)d_out;

  char* ws = (char*)d_ws;
  float* sc       = (float*)ws;                                   //  16 floats
  float* partials = (float*)(ws + 64);                            // 2048 floats
  h16*   xn  = (h16*)(ws + 16384);                                // 33.5M halfs
  h16*   wh  = (h16*)((char*)xn + (size_t)NX * 2);                // 1.3M halfs
  h16*   qkv = (h16*)((char*)wh + (size_t)NW * 2);                // 167.8M halfs
  h16*   s1  = (h16*)((char*)qkv + (size_t)B_ * N_ * N_ * FIVE_C * 2);
  h16*   s2  = (h16*)((char*)s1 + (size_t)NSLICE * 16384 * 2);

  reduce_partials_kernel<<<1024, 256, 0, stream>>>(x, partials);
  finalize_stats_kernel<<<1, 256, 0, stream>>>(partials, sc);
  norm_convert_kernel<<<(int)(NX / 4 / 256), 256, 0, stream>>>(x, sc, xn);
  convert_w_kernel<<<(int)(NW / 4 / 256), 256, 0, stream>>>(w, wh);
  qkv_gemm_kernel<<<dim3(B_ * N_ * N_ / 128, FIVE_C / 128), 256, 0, stream>>>(xn, wh, qkv);
  scores_kernel<<<NSLICE, 256, 0, stream>>>(qkv, s1, 1);
  scores_kernel<<<NSLICE, 256, 0, stream>>>(qkv, s2, 2);
  softmax_kernel<<<B_ * H_ * N_ * N_ / 8, 256, 0, stream>>>(s1, s2);
  out_kernel<<<NSLICE, 256, 0, stream>>>(qkv, s1, out, 1);
  out_kernel<<<NSLICE, 256, 0, stream>>>(qkv, s2, out, 2);
}
